// VariationalAttention_54460185313288
// MI455X (gfx1250) — matrix operand order repaired and checked by reading, not compiled, not run on hardware
//
#include <hip/hip_runtime.h>
#include <hip/hip_bf16.h>

#define BB 8
#define TT 256
#define SS 512
#define DD 512
#define EE 128
#define KK 8

typedef __attribute__((ext_vector_type(16))) __bf16        v16bf;
typedef __attribute__((ext_vector_type(2)))  __bf16        v2bf;
typedef __attribute__((ext_vector_type(8)))  float         v8f;
typedef __attribute__((ext_vector_type(8)))  unsigned int  v8u;

// pack two f32 -> packed bf16 (low = a, high = b)
__device__ __forceinline__ unsigned pack2bf(float a, float b) {
#if __has_builtin(__builtin_amdgcn_cvt_pk_bf16_f32)
  v2bf p = __builtin_amdgcn_cvt_pk_bf16_f32(a, b);
  return __builtin_bit_cast(unsigned, p);
#else
  // round-half-up then single v_perm_b32 packing the two high halves
  unsigned ua = __builtin_bit_cast(unsigned, a) + 0x8000u;
  unsigned ub = __builtin_bit_cast(unsigned, b) + 0x8000u;
  return __builtin_amdgcn_perm(ub, ua, 0x07060302u);
#endif
}

__device__ __forceinline__ uint2 pack4bf(float4 x) {
  uint2 r;
  r.x = pack2bf(x.x, x.y);
  r.y = pack2bf(x.z, x.w);
  return r;
}

__device__ __forceinline__ float fast_tanh(float x) {
  float cx = fminf(15.f, fmaxf(-15.f, x));
  float e  = __expf(2.f * cx);
  return 1.f - 2.f / (e + 1.f);
}

// Load a 16-bf16 fragment as two b128 LDS reads (p0/p1 must be 16B aligned).
__device__ __forceinline__ v16bf load_frag2(const unsigned short* p0,
                                            const unsigned short* p1) {
  v8u t;
  reinterpret_cast<uint4*>(&t)[0] = *reinterpret_cast<const uint4*>(p0);
  reinterpret_cast<uint4*>(&t)[1] = *reinterpret_cast<const uint4*>(p1);
  return __builtin_bit_cast(v16bf, t);
}

// ---------------------------------------------------------------------------
// Generic bf16-WMMA GEMM: C = act(A[M,K] x B[K,N] (+ Padd) (+ bias))
// 128 threads = 4 waves; block tile 128(M) x 64(N); K step 32.
// Double-buffered LDS; next tile's global loads are register-staged and issued
// before the current tile's 8 WMMAs so load latency hides behind matrix ops.
// A tile row-major (k-contiguous); B tile transposed (n-major): every
// fragment is two contiguous ds_load_b128.
// ---------------------------------------------------------------------------
__global__ __launch_bounds__(128)
void gemm_bf16_wmma(const float* __restrict__ A, const float* __restrict__ Bm,
                    float* __restrict__ C, const float* __restrict__ Padd,
                    const float* __restrict__ bias,
                    int M, int N, int Kd, int lda, int ldb, int ldc,
                    long long bsA, long long bsB, long long bsC, int doTanh) {
  __shared__ unsigned short As[2][128][40];  // 2 x (128 m-rows x 32 k + pad)
  __shared__ unsigned short Bs[2][64][40];   // 2 x (64 n-rows x 32 k + pad)

  const int tid  = threadIdx.x;
  const int wave = tid >> 5;
  const int lane = tid & 31;
  const int half = lane >> 4;        // 0: lanes 0-15, 1: lanes 16-31
  const int lr   = lane & 15;
  const int n0   = blockIdx.x * 64;
  const int m0   = blockIdx.y * 128;

  const float* Ab = A  + (long long)blockIdx.z * bsA;
  const float* Bb = Bm + (long long)blockIdx.z * bsB;
  float*       Cb = C  + (long long)blockIdx.z * bsC;

  const int nl = tid & 63;        // B staging: column owned by this thread
  const int kq = tid >> 6;        // 0/1: which k-quads within each group of 8

  float4 ra[8];                   // register-staged A tile slice
  float  rb[16];                  // register-staged B tile slice

  auto loadTileRegs = [&](int k0) {
#pragma unroll
    for (int it = 0; it < 8; ++it) {
      int fi  = tid + it * 128;          // 0..1023 float4s of the A tile
      int row = fi >> 3;
      int kc  = (fi & 7) << 2;
      ra[it] = *reinterpret_cast<const float4*>(
          Ab + (long long)(m0 + row) * lda + (k0 + kc));
    }
#pragma unroll
    for (int g = 0; g < 4; ++g) {
      int kk = kq * 4 + g * 8;           // 0..31
#pragma unroll
      for (int j = 0; j < 4; ++j)
        rb[g * 4 + j] = Bb[(long long)(k0 + kk + j) * ldb + (n0 + nl)];
    }
  };

  auto storeTileLds = [&](int buf) {
#pragma unroll
    for (int it = 0; it < 8; ++it) {
      int fi  = tid + it * 128;
      int row = fi >> 3;
      int kc  = (fi & 7) << 2;
      *reinterpret_cast<uint2*>(&As[buf][row][kc]) = pack4bf(ra[it]);
    }
#pragma unroll
    for (int g = 0; g < 4; ++g) {
      int   kk = kq * 4 + g * 8;
      uint2 pk;
      pk.x = pack2bf(rb[g * 4 + 0], rb[g * 4 + 1]);
      pk.y = pack2bf(rb[g * 4 + 2], rb[g * 4 + 3]);
      *reinterpret_cast<uint2*>(&Bs[buf][nl][kk]) = pk;
    }
  };

  v8f acc[2][4];
#pragma unroll
  for (int mm = 0; mm < 2; ++mm)
#pragma unroll
    for (int t = 0; t < 4; ++t)
#pragma unroll
      for (int v = 0; v < 8; ++v) acc[mm][t][v] = 0.f;

  // prologue: stage tile 0
  loadTileRegs(0);
  storeTileLds(0);
  __syncthreads();

  int cur = 0;
  for (int k0 = 0; k0 < Kd; k0 += 32) {
    const bool more = (k0 + 32) < Kd;
    if (more) {
      loadTileRegs(k0 + 32);             // issue loads; no wait needed yet
      if (k0 + 64 < Kd) {                // keep L2 warm (global_prefetch_b8)
        __builtin_prefetch(Ab + (long long)(m0 + tid) * lda + (k0 + 64), 0, 0);
        __builtin_prefetch(Bb + (long long)(k0 + 64 + (tid & 31)) * ldb + n0, 0, 0);
      }
    }

    // ---- compute current tile: contiguous 16B chunks -> ds_load_b128 ----
    // A (ISA 7.12.2): lane half h, k in [8h,8h+8) U [8h+16,8h+24)
    const int kb = half << 3;
    v16bf afrag[2];
#pragma unroll
    for (int mm = 0; mm < 2; ++mm) {
      const unsigned short* ar = &As[cur][wave * 32 + mm * 16 + lr][0];
      afrag[mm] = load_frag2(ar + kb, ar + kb + 16);
    }
#pragma unroll
    for (int t = 0; t < 4; ++t) {
      // B: lane holds col (t*16+lr), k in [8h,8h+8) U [8h+16,8h+24)
      const unsigned short* br = &Bs[cur][t * 16 + lr][half << 3];
      v16bf bfrag = load_frag2(br, br + 16);
#pragma unroll
      for (int mm = 0; mm < 2; ++mm)
        acc[mm][t] = __builtin_amdgcn_wmma_f32_16x16x32_bf16(
            false, afrag[mm], false, bfrag, (short)0, acc[mm][t], false, false);
    }

    if (more) storeTileLds(cur ^ 1);     // waits loads, converts, fills alt buf
    __syncthreads();
    cur ^= 1;
  }

  // ---- epilogue: C/D layout -> VGPR v is row (v + 8*half), col = lr ----
#pragma unroll
  for (int mm = 0; mm < 2; ++mm)
#pragma unroll
    for (int t = 0; t < 4; ++t) {
      int   n  = n0 + t * 16 + lr;
      float bv = bias ? bias[n] : 0.f;
#pragma unroll
      for (int v = 0; v < 8; ++v) {
        int   m   = m0 + wave * 32 + mm * 16 + half * 8 + v;
        float val = acc[mm][t][v];
        if (Padd) val += Padd[(long long)m * ldc + n];
        val += bv;
        if (doTanh) val = fast_tanh(val);
        Cb[(long long)m * ldc + n] = val;
      }
    }
}

// ---------------------------------------------------------------------------
// scores[b,t,s] = sum_e v[e]*tanh(wq[b,t,e]+uh[b,s,e]); masked log-softmax.
// One block handles 16 t-rows of one batch; wq/v cached in LDS.
// ---------------------------------------------------------------------------
__global__ __launch_bounds__(256)
void scores_softmax_kernel(const float* __restrict__ wq, const float* __restrict__ uh,
                           const float* __restrict__ vvec, const int* __restrict__ mlen,
                           float* __restrict__ alpha, float* __restrict__ log_alpha) {
  __shared__ float wq_s[16][EE];
  __shared__ float v_s[EE];
  __shared__ float sc[16][SS];
  __shared__ float red[256];

  const int tid = threadIdx.x;
  const int b   = blockIdx.x / (TT / 16);
  const int t0  = (blockIdx.x % (TT / 16)) * 16;

  for (int i = tid; i < 16 * EE; i += 256) {
    int r = i / EE, c = i % EE;
    wq_s[r][c] = wq[((long long)(b * TT + t0 + r)) * EE + c];
  }
  if (tid < EE) v_s[tid] = vvec[tid];
  __syncthreads();

#pragma unroll
  for (int pass = 0; pass < 2; ++pass) {
    int s = tid + pass * 256;
    const float* uhr = uh + ((long long)(b * SS + s)) * EE;
    float acc[16];
#pragma unroll
    for (int tl = 0; tl < 16; ++tl) acc[tl] = 0.f;
    for (int e = 0; e < EE; ++e) {
      float u  = uhr[e];
      float ve = v_s[e];
#pragma unroll
      for (int tl = 0; tl < 16; ++tl)
        acc[tl] = fmaf(ve, fast_tanh(wq_s[tl][e] + u), acc[tl]);
    }
#pragma unroll
    for (int tl = 0; tl < 16; ++tl) sc[tl][s] = acc[tl];
  }
  __syncthreads();

  const int len = mlen[b];
  for (int i = tid; i < 16 * SS; i += 256) {
    int tl = i >> 9, s = i & 511;
    if (s >= len) sc[tl][s] = -__builtin_inff();
  }
  __syncthreads();

  for (int tl = 0; tl < 16; ++tl) {
    float x0 = sc[tl][tid], x1 = sc[tl][tid + 256];
    red[tid] = fmaxf(x0, x1);
    __syncthreads();
    for (int off = 128; off > 0; off >>= 1) {
      if (tid < off) red[tid] = fmaxf(red[tid], red[tid + off]);
      __syncthreads();
    }
    float mx = red[0];
    __syncthreads();
    red[tid] = __expf(x0 - mx) + __expf(x1 - mx);
    __syncthreads();
    for (int off = 128; off > 0; off >>= 1) {
      if (tid < off) red[tid] += red[tid + off];
      __syncthreads();
    }
    float lz = __logf(red[0]);
    __syncthreads();
    long long base = ((long long)(b * TT + t0 + tl)) * SS;
    float la0 = x0 - mx - lz, la1 = x1 - mx - lz;
    log_alpha[base + tid]       = la0;
    log_alpha[base + tid + 256] = la1;
    alpha[base + tid]           = __expf(la0);
    alpha[base + tid + 256]     = __expf(la1);
  }
}

// ---------------------------------------------------------------------------
// Deterministic Gumbel-max sampling: one wave32 per (k,b,t) sample.
// ---------------------------------------------------------------------------
__device__ __forceinline__ unsigned hash32(unsigned x) {
  x ^= x >> 16; x *= 0x7FEB352Du;
  x ^= x >> 15; x *= 0x846CA68Bu;
  x ^= x >> 16;
  return x;
}

__global__ __launch_bounds__(128)
void sample_kernel(const float* __restrict__ log_alpha, int* __restrict__ idx_out,
                   float* __restrict__ slp_out) {
  const int wave = threadIdx.x >> 5, lane = threadIdx.x & 31;
  const int sid  = blockIdx.x * 4 + wave;   // 0..K*B*T-1, k-major
  const int bt   = sid % (BB * TT);
  const int b    = bt / TT, t = bt % TT;
  const float* la = log_alpha + ((long long)(b * TT + t)) * SS;

  float bestv = -__builtin_inff();
  int   besti = 0;
#pragma unroll
  for (int j = 0; j < 16; ++j) {
    int s = lane + j * 32;
    unsigned h = hash32((unsigned)(sid * SS + s) ^ 0xA5C42042u);
    float u = (float)(h >> 8) * (1.f / 16777216.f) + (0.5f / 16777216.f);
    float g = -__logf(-__logf(u));
    float val = la[s] + g;
    if (val > bestv) { bestv = val; besti = s; }
  }
#pragma unroll
  for (int off = 16; off > 0; off >>= 1) {
    float ov = __shfl_xor(bestv, off, 32);
    int   oi = __shfl_xor(besti, off, 32);
    if (ov > bestv || (ov == bestv && oi < besti)) { bestv = ov; besti = oi; }
  }
  if (lane == 0) {
    idx_out[sid] = besti;
    slp_out[sid] = la[besti];
  }
}

// ---------------------------------------------------------------------------
// h_y[k,b,t,:] = tanh(P[b,t,:] + M2[b, idx[k,b,t], :] + bout)
// ---------------------------------------------------------------------------
__global__ __launch_bounds__(128)
void hy_kernel(const float* __restrict__ P, const float* __restrict__ M2,
               const float* __restrict__ bout, const int* __restrict__ idx,
               float* __restrict__ hy) {
  const long long r  = blockIdx.x;            // 0..K*B*T-1
  const int       bt = (int)(r % (BB * TT));
  const int       b  = bt / TT;
  const int       id = idx[r];
  const float4* p4 = reinterpret_cast<const float4*>(P + (long long)bt * DD);
  const float4* m4 = reinterpret_cast<const float4*>(M2 + ((long long)(b * SS + id)) * DD);
  const float4* b4 = reinterpret_cast<const float4*>(bout);
  float4*       o4 = reinterpret_cast<float4*>(hy + r * DD);
  const int c = threadIdx.x;                  // 128 threads x float4 = 512
  float4 pv = p4[c], mv = m4[c], bv = b4[c], ov;
  ov.x = fast_tanh(pv.x + mv.x + bv.x);
  ov.y = fast_tanh(pv.y + mv.y + bv.y);
  ov.z = fast_tanh(pv.z + mv.z + bv.z);
  ov.w = fast_tanh(pv.w + mv.w + bv.w);
  o4[c] = ov;
}

// ---------------------------------------------------------------------------
extern "C" void kernel_launch(void* const* d_in, const int* in_sizes, int n_in,
                              void* d_out, int out_size, void* d_ws, size_t ws_size,
                              hipStream_t stream) {
  const float* input = (const float*)d_in[0];   // (B,T,512)
  const float* mem   = (const float*)d_in[1];   // (B,S,512)
  const int*   mlen  = (const int*)d_in[2];     // (B,)
  const float* Wq    = (const float*)d_in[3];   // (512,128)
  const float* Wc    = (const float*)d_in[4];   // (512,128)
  const float* vvec  = (const float*)d_in[5];   // (128,)
  const float* Wout  = (const float*)d_in[6];   // (1024,512)
  const float* bout  = (const float*)d_in[7];   // (512,)

  // output layout (flat concat): h_c, h_y, alpha, context_c, sample_log_probs
  float* out   = (float*)d_out;
  float* h_c   = out;                 // 2048*512
  float* h_y   = out + 1048576;       // 16384*512
  float* alpha = out + 9437184;       // 2048*512
  float* ctx_c = out + 10485760;      // 2048*512
  float* slp   = out + 11534336;      // 16384

  // workspace
  float* ws  = (float*)d_ws;
  float* wq  = ws;                    // 2048*128
  float* uh  = ws + 262144;           // 4096*128
  float* lga = ws + 786432;           // 2048*512  log_alpha
  float* Pb  = ws + 1835008;          // 2048*512  input @ Wout_top
  float* M2  = ws + 2883584;          // 4096*512  mem @ Wout_bot
  int*   idx = (int*)(ws + 4980736);  // 16384 ints

  const float* Wout_bot = Wout + 512 * 512;
  dim3 blk(128);

  // 1) wq = input @ Wq             (2048 x 128, K=512)
  gemm_bf16_wmma<<<dim3(2, 16, 1), blk, 0, stream>>>(
      input, Wq, wq, nullptr, nullptr, 2048, 128, 512, 512, 128, 128, 0, 0, 0, 0);
  // 2) uh = mem @ Wc               (4096 x 128, K=512)
  gemm_bf16_wmma<<<dim3(2, 32, 1), blk, 0, stream>>>(
      mem, Wc, uh, nullptr, nullptr, 4096, 128, 512, 512, 128, 128, 0, 0, 0, 0);
  // 3) scores + masked log-softmax -> alpha (out), log_alpha (ws)
  scores_softmax_kernel<<<dim3(BB * (TT / 16)), dim3(256), 0, stream>>>(
      wq, uh, vvec, mlen, alpha, lga);
  // 4) context_c[b] = alpha[b] @ mem[b]   (batched 256 x 512, K=512)
  gemm_bf16_wmma<<<dim3(8, 2, BB), blk, 0, stream>>>(
      alpha, mem, ctx_c, nullptr, nullptr, 256, 512, 512, 512, 512, 512,
      (long long)TT * SS, (long long)SS * DD, (long long)TT * DD, 0);
  // 5) P = input @ Wout_top        (2048 x 512, K=512)
  gemm_bf16_wmma<<<dim3(8, 16, 1), blk, 0, stream>>>(
      input, Wout, Pb, nullptr, nullptr, 2048, 512, 512, 512, 512, 512, 0, 0, 0, 0);
  // 6) M2 = mem @ Wout_bot         (4096 x 512, K=512)
  gemm_bf16_wmma<<<dim3(8, 32, 1), blk, 0, stream>>>(
      mem, Wout_bot, M2, nullptr, nullptr, 4096, 512, 512, 512, 512, 512, 0, 0, 0, 0);
  // 7) h_c = tanh(ctx_c @ Wout_bot + P + bout)
  gemm_bf16_wmma<<<dim3(8, 16, 1), blk, 0, stream>>>(
      ctx_c, Wout_bot, h_c, Pb, bout, 2048, 512, 512, 512, 512, 512, 0, 0, 0, 1);
  // 8) Gumbel-max sampling -> idx, sample_log_probs
  sample_kernel<<<dim3(KK * BB * TT / 4), blk, 0, stream>>>(lga, idx, slp);
  // 9) h_y = tanh(P + M2[gather] + bout)
  hy_kernel<<<dim3(KK * BB * TT), blk, 0, stream>>>(Pb, M2, bout, idx, h_y);
}
